// MatchAttnMoEDeep_34608846471455
// MI455X (gfx1250) — matrix operand order repaired and checked by reading, not compiled, not run on hardware
//
#include <hip/hip_runtime.h>
#include <hip/hip_bf16.h>

typedef __bf16 bf16_t;
typedef __attribute__((ext_vector_type(16))) __bf16 v16bf;
typedef __attribute__((ext_vector_type(8)))  float  v8f;
typedef int vi4 __attribute__((vector_size(16)));

#define B_TOT 16384
#define S_LEN 10
#define F_IN  13
#define D_DIM 64
#define H_CNT 16
#define L_CNT 16
#define EPSF  1e-5f

#if defined(__HIP_DEVICE_COMPILE__) && \
    __has_builtin(__builtin_amdgcn_global_load_async_to_lds_b128)
#define HAVE_ASYNC_LDS 1
#else
#define HAVE_ASYNC_LDS 0
#endif

__device__ __forceinline__ bf16_t to_bf(float x) { return (bf16_t)x; }

__device__ __forceinline__ v8f zero8() {
  v8f z;
#pragma unroll
  for (int i = 0; i < 8; ++i) z[i] = 0.f;
  return z;
}

__device__ __forceinline__ v8f wmma_bf16(v16bf a, v16bf b, v8f c) {
  return __builtin_amdgcn_wmma_f32_16x16x32_bf16(false, a, false, b, (short)0, c,
                                                 false, false);
}

// 16-byte global -> LDS copy; async path on toolchains that expose it.
__device__ __forceinline__ void cp16_g2l(const bf16_t* gsrc, bf16_t* ldst) {
#if HAVE_ASYNC_LDS
  __builtin_amdgcn_global_load_async_to_lds_b128(
      (__attribute__((address_space(1))) vi4*)(void*)gsrc,
      (__attribute__((address_space(3))) vi4*)(void*)ldst, 0, 0);
#else
  *(uint4*)ldst = *(const uint4*)gsrc;
#endif
}
__device__ __forceinline__ void cp_wait() {
#if HAVE_ASYNC_LDS
  asm volatile("s_wait_asynccnt 0x0" ::: "memory");
#endif
}

// A fragment (16x32 bf16): lane -> M = lane%16; lanes 0-15 hold K=0..15, lanes 16-31 hold K=16..31.
__device__ __forceinline__ v16bf load_a_lds(const bf16_t* X, int stride, int row0,
                                            int k0, int lane) {
  int m  = lane & 15;
  int hi = (lane & 16) ? 16 : 0;
  const bf16_t* p = X + (size_t)(row0 + m) * stride + k0 + hi;
  v16bf f;
#pragma unroll
  for (int v = 0; v < 8; ++v) {
    int kk = 2 * v;
    f[2 * v]     = p[kk];
    f[2 * v + 1] = p[kk + 1];
  }
  return f;
}

// B fragment (32x16 bf16) where B[k][n] = X[n_row0+n][k0+k]  (contiguous along k):
// lanes 0-15 hold K=0..15, lanes 16-31 hold K=16..31, N = lane%16.
__device__ __forceinline__ v16bf load_bT(const bf16_t* X, int stride, int n_row0,
                                         int k0, int lane) {
  const bf16_t* p = X + (size_t)(n_row0 + (lane & 15)) * stride + k0 +
                    ((lane & 16) ? 16 : 0);
  v16bf f;
#pragma unroll
  for (int v = 0; v < 16; ++v) f[v] = p[v];
  return f;
}

// B fragment (32x16 bf16) where B[k][n] = X[k_row0+k][n0+n] (column gather).
__device__ __forceinline__ v16bf load_b_col(const bf16_t* X, int stride, int k_row0,
                                            int n0, int lane) {
  int n  = n0 + (lane & 15);
  int kb = (lane & 16) ? 16 : 0;
  v16bf f;
#pragma unroll
  for (int v = 0; v < 8; ++v) {
    f[2 * v]     = X[(size_t)(k_row0 + kb + 2 * v) * stride + n];
    f[2 * v + 1] = X[(size_t)(k_row0 + kb + 2 * v + 1) * stride + n];
  }
  return f;
}

// ---------------------------------------------------------------- utility kernels
__global__ void k_cast(const float* __restrict__ src, bf16_t* __restrict__ dst,
                       int n) {
  int i = blockIdx.x * blockDim.x + threadIdx.x;
  if (i < n) dst[i] = (bf16_t)src[i];
}

__global__ void k_zero(float* __restrict__ p, int n) {
  int i = blockIdx.x * blockDim.x + threadIdx.x;
  if (i < n) p[i] = 0.f;
}

// per-channel sum / sumsq with LDS pre-reduction; ch = (i/ndiv) % nch
__global__ __launch_bounds__(256) void k_colstats(const float* __restrict__ x,
                                                  long n, int ndiv, int nch,
                                                  float* __restrict__ sum,
                                                  float* __restrict__ sumsq) {
  __shared__ float ls[512];
  __shared__ float lq[512];
  for (int j = threadIdx.x; j < nch; j += blockDim.x) { ls[j] = 0.f; lq[j] = 0.f; }
  __syncthreads();
  long i = (long)blockIdx.x * blockDim.x + threadIdx.x;
  long stride = (long)gridDim.x * blockDim.x;
  for (; i < n; i += stride) {
    float v = x[i];
    int ch  = (int)((i / ndiv) % nch);
    atomicAdd(&ls[ch], v);
    atomicAdd(&lq[ch], v * v);
  }
  __syncthreads();
  for (int j = threadIdx.x; j < nch; j += blockDim.x) {
    atomicAdd(&sum[j], ls[j]);
    atomicAdd(&sumsq[j], lq[j]);
  }
}

// BN over S channels of x[B,S,F] then linear F->D : h[b,s,d]
__global__ void k_embed(const float* __restrict__ x, const float* __restrict__ st,
                        const float* __restrict__ g, const float* __restrict__ b,
                        const float* __restrict__ Wi, const float* __restrict__ bi,
                        float* __restrict__ h) {
  int i = blockIdx.x * blockDim.x + threadIdx.x;
  if (i >= B_TOT * S_LEN * D_DIM) return;
  int d     = i & 63;
  long row  = i >> 6;          // b*10 + s
  int s     = (int)(row % S_LEN);
  float cnt = (float)B_TOT * (float)F_IN;
  float mean = st[s] / cnt;
  float var  = st[16 + s] / cnt - mean * mean;
  float sc   = g[s] * rsqrtf(var + EPSF);
  float sh   = b[s] - mean * sc;
  const float* xr = x + row * F_IN;
  const float* wr = Wi + d * F_IN;
  float acc = bi[d];
#pragma unroll
  for (int f = 0; f < F_IN; ++f) acc += (xr[f] * sc + sh) * wr[f];
  h[i] = acc;
}

// ------------------------------------------------------- persistent fused 16 layers
__global__ __launch_bounds__(256) void k_transformer(
    float* __restrict__ hio, const bf16_t* __restrict__ wq,
    const bf16_t* __restrict__ wk, const bf16_t* __restrict__ wv,
    const bf16_t* __restrict__ wp, const bf16_t* __restrict__ w1,
    const bf16_t* __restrict__ w2, const float* __restrict__ bq,
    const float* __restrict__ bk, const float* __restrict__ bvv,
    const float* __restrict__ wg, const float* __restrict__ bg,
    const float* __restrict__ bp, const float* __restrict__ b1,
    const float* __restrict__ b2, const float* __restrict__ ln1g,
    const float* __restrict__ ln1b, const float* __restrict__ ln2g,
    const float* __restrict__ ln2b) {
  extern __shared__ __align__(16) char smem_raw[];
  bf16_t* Xb  = (bf16_t*)smem_raw;   // [160][72]
  bf16_t* Qb  = Xb + 160 * 72;       // [192][72]  (rows 160..191 zero pad)
  bf16_t* Kb  = Qb + 192 * 72;       // [192][72]
  bf16_t* Vb  = Kb + 192 * 72;       // [192][72]
  bf16_t* HOb = Vb + 192 * 72;       // [160][72]  (reused as FFN hidden chunk)
  bf16_t* ATT = HOb + 160 * 72;      // [16][16][32] zero padded
  float* SC = (float*)(ATT + 16 * 16 * 32);  // [16][16][17]
  float* XF = SC + 16 * 16 * 17;             // [160][66]
  float* XM = XF + 160 * 66;                 // [16][64]
  float* GT = XM + 16 * 64;                  // [256]
  bf16_t* WS = (bf16_t*)(GT + 256);          // [4][64][72] staged weights

  const int tid  = threadIdx.x;
  const int lane = tid & 31;
  const int wv_id = tid >> 5;  // wave id 0..7
  const int b0 = blockIdx.x * 16;

  // load activation tile, fp32 + bf16 mirror
  for (int i = tid; i < 160 * 64; i += 256) {
    int r = i >> 6, c = i & 63;
    float v = hio[(size_t)(b0 + r / S_LEN) * 640 + (r % S_LEN) * 64 + c];
    XF[r * 66 + c] = v;
    Xb[r * 72 + c] = to_bf(v);
  }
  // zero padded Q/K/V rows and attention scratch (keeps padded WMMA tiles NaN-free)
  for (int i = tid; i < 32 * 72; i += 256) {
    Qb[160 * 72 + i] = to_bf(0.f);
    Kb[160 * 72 + i] = to_bf(0.f);
    Vb[160 * 72 + i] = to_bf(0.f);
  }
  for (int i = tid; i < 16 * 16 * 32; i += 256) ATT[i] = to_bf(0.f);
  __syncthreads();

  for (int li = 0; li < L_CNT; ++li) {
    // ------- gate: softmax_h( mean_s(x) . Wg[h] + bg[h] )
    for (int i = tid; i < 16 * 64; i += 256) {
      int bb = i >> 6, d = i & 63;
      float s = 0.f;
      for (int ss = 0; ss < S_LEN; ++ss) s += XF[(bb * S_LEN + ss) * 66 + d];
      XM[i] = s * 0.1f;
    }
    __syncthreads();
    {
      int bb = tid >> 4, hh = tid & 15;
      const float* wgp = wg + ((size_t)li * 16 + hh) * 64;
      float acc = bg[li * 16 + hh];
      for (int d = 0; d < 64; ++d) acc += XM[bb * 64 + d] * wgp[d];
      GT[tid] = acc;
    }
    __syncthreads();
    if (tid < 16) {
      float mx = -1e30f;
      for (int hh = 0; hh < 16; ++hh) mx = fmaxf(mx, GT[tid * 16 + hh]);
      float sm = 0.f;
      for (int hh = 0; hh < 16; ++hh) {
        float e = expf(GT[tid * 16 + hh] - mx);
        GT[tid * 16 + hh] = e;
        sm += e;
      }
      float inv = 1.f / sm;
      for (int hh = 0; hh < 16; ++hh) GT[tid * 16 + hh] *= inv;
    }
    __syncthreads();

    v8f accP[5];
#pragma unroll
    for (int i = 0; i < 5; ++i) accP[i] = zero8();

    for (int hh = 0; hh < H_CNT; ++hh) {
      const size_t wofs = ((size_t)li * 16 + hh) * 64 * 64;

      // prefetch next head's weights (one lane; lowers to global_prefetch_b8)
      if (tid == 0 && hh + 1 < H_CNT) {
        const size_t nofs = ((size_t)li * 16 + hh + 1) * 64 * 64;
        __builtin_prefetch(wq + nofs, 0, 1);
        __builtin_prefetch(wk + nofs, 0, 1);
        __builtin_prefetch(wv + nofs, 0, 1);
        __builtin_prefetch(wp + (size_t)li * 65536 + (hh + 1) * 64, 0, 1);
      }

      // -------- stage per-head weights into LDS: WS[0..2]=Wq/Wk/Wv, WS[3]=Wp slice
      for (int c = tid; c < 2048; c += 256) {
        int mat = c >> 9;
        int rem = c & 511;
        int row = rem >> 3;
        int k8  = (rem & 7) * 8;
        const bf16_t* s;
        if (mat == 0)      s = wq + wofs + (size_t)row * 64 + k8;
        else if (mat == 1) s = wk + wofs + (size_t)row * 64 + k8;
        else if (mat == 2) s = wv + wofs + (size_t)row * 64 + k8;
        else               s = wp + (size_t)li * 65536 + (size_t)row * 1024 +
                               hh * 64 + k8;
        cp16_g2l(s, WS + mat * 64 * 72 + row * 72 + k8);
      }
      cp_wait();
      __syncthreads();

      // -------- QKV: 120 tiles (3 mats x 10 Mtiles x 4 Ntiles), 15 per wave
      for (int i = 0; i < 15; ++i) {
        int t = wv_id * 15 + i;
        int mat = t / 40, rem = t % 40;
        int mt = rem >> 2, nt = rem & 3;
        const bf16_t* Wst = WS + mat * 64 * 72;
        const float* bia = (mat == 0) ? bq : ((mat == 1) ? bk : bvv);
        v8f acc = zero8();
#pragma unroll
        for (int ks = 0; ks < 64; ks += 32) {
          v16bf a = load_a_lds(Xb, 72, mt * 16, ks, lane);
          v16bf bfr = load_bT(Wst, 72, nt * 16, ks, lane);  // B[k][n]=W[n][k]
          acc = wmma_bf16(a, bfr, acc);
        }
        bf16_t* dst = (mat == 0) ? Qb : ((mat == 1) ? Kb : Vb);
        int n = lane & 15, hi8 = (lane & 16) ? 8 : 0;
        float bval = bia[((size_t)li * 16 + hh) * 64 + nt * 16 + n];
#pragma unroll
        for (int v = 0; v < 8; ++v)
          dst[(mt * 16 + v + hi8) * 72 + nt * 16 + n] = to_bf(acc[v] + bval);
      }
      __syncthreads();
      // -------- scores = Q K^T / 8 : 2 batches per wave
      for (int i = 0; i < 2; ++i) {
        int bb = wv_id * 2 + i;
        v8f acc = zero8();
#pragma unroll
        for (int ks = 0; ks < 64; ks += 32) {
          v16bf a = load_a_lds(Qb, 72, bb * S_LEN, ks, lane);
          v16bf bfr = load_bT(Kb, 72, bb * S_LEN, ks, lane);
          acc = wmma_bf16(a, bfr, acc);
        }
        int n = lane & 15, hi8 = (lane & 16) ? 8 : 0;
#pragma unroll
        for (int v = 0; v < 8; ++v) SC[(bb * 16 + v + hi8) * 17 + n] = acc[v] * 0.125f;
      }
      __syncthreads();
      // -------- softmax rows (one (b,s) row per lane)
      if (tid < 160) {
        int bb = tid / S_LEN, ss = tid % S_LEN;
        float* row = SC + (bb * 16 + ss) * 17;
        float mx = -1e30f;
        for (int tt = 0; tt < S_LEN; ++tt) mx = fmaxf(mx, row[tt]);
        float e[S_LEN];
        float sm = 0.f;
        for (int tt = 0; tt < S_LEN; ++tt) { e[tt] = expf(row[tt] - mx); sm += e[tt]; }
        float inv = 1.f / sm;
        bf16_t* arow = ATT + (bb * 16 + ss) * 32;
        for (int tt = 0; tt < S_LEN; ++tt) arow[tt] = to_bf(e[tt] * inv);
      }
      __syncthreads();
      // -------- ho = attn @ V  (scaled by gate): 64 jobs, 8 per wave
      for (int i = 0; i < 8; ++i) {
        int j = wv_id * 8 + i;
        int bb = j >> 2, nt = j & 3;
        v16bf a = load_a_lds(ATT + bb * 16 * 32, 32, 0, 0, lane);
        v16bf bfr = load_b_col(Vb, 72, bb * S_LEN, nt * 16, lane);
        v8f acc = wmma_bf16(a, bfr, zero8());
        float gte = GT[bb * 16 + hh];
        int n = lane & 15, hi8 = (lane & 16) ? 8 : 0;
#pragma unroll
        for (int v = 0; v < 8; ++v) {
          int m = v + hi8;
          if (m < S_LEN) HOb[(bb * S_LEN + m) * 72 + nt * 16 + n] = to_bf(acc[v] * gte);
        }
      }
      __syncthreads();
      // -------- output proj: accumulate ho_h @ Wp[:, h*64:(h+1)*64]^T (staged)
      for (int i = 0; i < 5; ++i) {
        int t = wv_id * 5 + i;
        int mt = t >> 2, nt = t & 3;
#pragma unroll
        for (int ks = 0; ks < 64; ks += 32) {
          v16bf a = load_a_lds(HOb, 72, mt * 16, ks, lane);
          v16bf bfr = load_bT(WS + 3 * 64 * 72, 72, nt * 16, ks, lane);
          accP[i] = wmma_bf16(a, bfr, accP[i]);
        }
      }
      __syncthreads();
    }  // heads

    // proj bias + residual -> XF
    for (int i = 0; i < 5; ++i) {
      int t = wv_id * 5 + i;
      int mt = t >> 2, nt = t & 3;
      int n = lane & 15, hi8 = (lane & 16) ? 8 : 0;
      float bpv = bp[li * 64 + nt * 16 + n];
#pragma unroll
      for (int v = 0; v < 8; ++v)
        XF[(mt * 16 + v + hi8) * 66 + nt * 16 + n] += accP[i][v] + bpv;
    }
    __syncthreads();
    // LN1
    if (tid < 160) {
      float m = 0.f, s2 = 0.f;
      for (int d = 0; d < 64; ++d) { float v = XF[tid * 66 + d]; m += v; s2 += v * v; }
      m *= (1.f / 64.f);
      s2 = s2 * (1.f / 64.f) - m * m;
      float inv = rsqrtf(s2 + EPSF);
      for (int d = 0; d < 64; ++d) {
        float v = (XF[tid * 66 + d] - m) * inv * ln1g[li * 64 + d] + ln1b[li * 64 + d];
        XF[tid * 66 + d] = v;
        Xb[tid * 72 + d] = to_bf(v);
      }
    }
    __syncthreads();

    // -------- FFN: D -> 256 (relu) -> D, hidden tiled in chunks of 64
    v8f acc2[5];
#pragma unroll
    for (int i = 0; i < 5; ++i) acc2[i] = zero8();
    for (int c = 0; c < 4; ++c) {
      // stage W1 chunk (rows c*64..c*64+63) and W2 chunk (cols c*64..) into LDS
      for (int cc = tid; cc < 1024; cc += 256) {
        int mat = cc >> 9;
        int rem = cc & 511;
        int row = rem >> 3;
        int k8  = (rem & 7) * 8;
        const bf16_t* s =
            mat ? (w2 + (size_t)li * 64 * 256 + (size_t)row * 256 + c * 64 + k8)
                : (w1 + ((size_t)li * 256 + c * 64 + row) * 64 + k8);
        cp16_g2l(s, WS + mat * 64 * 72 + row * 72 + k8);
      }
      cp_wait();
      __syncthreads();
      for (int i = 0; i < 5; ++i) {
        int t = wv_id * 5 + i;
        int mt = t >> 2, nt = t & 3;
        v8f acc = zero8();
#pragma unroll
        for (int ks = 0; ks < 64; ks += 32) {
          v16bf a = load_a_lds(Xb, 72, mt * 16, ks, lane);
          v16bf bfr = load_bT(WS, 72, nt * 16, ks, lane);
          acc = wmma_bf16(a, bfr, acc);
        }
        int n = lane & 15, hi8 = (lane & 16) ? 8 : 0;
        float b1v = b1[li * 256 + c * 64 + nt * 16 + n];
#pragma unroll
        for (int v = 0; v < 8; ++v)
          HOb[(mt * 16 + v + hi8) * 72 + nt * 16 + n] = to_bf(fmaxf(acc[v] + b1v, 0.f));
      }
      __syncthreads();
      for (int i = 0; i < 5; ++i) {
        int t = wv_id * 5 + i;
        int mt = t >> 2, nt = t & 3;
#pragma unroll
        for (int ks = 0; ks < 64; ks += 32) {
          v16bf a = load_a_lds(HOb, 72, mt * 16, ks, lane);
          v16bf bfr = load_bT(WS + 64 * 72, 72, nt * 16, ks, lane);
          acc2[i] = wmma_bf16(a, bfr, acc2[i]);
        }
      }
      __syncthreads();
    }
    // FFN bias + residual
    for (int i = 0; i < 5; ++i) {
      int t = wv_id * 5 + i;
      int mt = t >> 2, nt = t & 3;
      int n = lane & 15, hi8 = (lane & 16) ? 8 : 0;
      float b2v = b2[li * 64 + nt * 16 + n];
#pragma unroll
      for (int v = 0; v < 8; ++v)
        XF[(mt * 16 + v + hi8) * 66 + nt * 16 + n] += acc2[i][v] + b2v;
    }
    __syncthreads();
    // LN2
    if (tid < 160) {
      float m = 0.f, s2 = 0.f;
      for (int d = 0; d < 64; ++d) { float v = XF[tid * 66 + d]; m += v; s2 += v * v; }
      m *= (1.f / 64.f);
      s2 = s2 * (1.f / 64.f) - m * m;
      float inv = rsqrtf(s2 + EPSF);
      for (int d = 0; d < 64; ++d) {
        float v = (XF[tid * 66 + d] - m) * inv * ln2g[li * 64 + d] + ln2b[li * 64 + d];
        XF[tid * 66 + d] = v;
        Xb[tid * 72 + d] = to_bf(v);
      }
    }
    __syncthreads();
  }  // layers

  for (int i = tid; i < 160 * 64; i += 256) {
    int r = i >> 6, c = i & 63;
    hio[(size_t)(b0 + r / S_LEN) * 640 + (r % S_LEN) * 64 + c] = XF[r * 66 + c];
  }
}

// BN over S of final h, write normalized bf16 flat [B,640]
__global__ void k_bnout(const float* __restrict__ h, const float* __restrict__ st,
                        const float* __restrict__ g, const float* __restrict__ b,
                        bf16_t* __restrict__ hn) {
  int i = blockIdx.x * blockDim.x + threadIdx.x;
  if (i >= B_TOT * 640) return;
  int s = (i >> 6) % S_LEN;
  float cnt  = (float)B_TOT * 64.f;
  float mean = st[32 + s] / cnt;
  float var  = st[48 + s] / cnt - mean * mean;
  float sc   = g[s] * rsqrtf(var + EPSF);
  float sh   = b[s] - mean * sc;
  hn[i] = (bf16_t)(h[i] * sc + sh);
}

// FC1: z[B,512] = hn[B,640] @ Wf1^T + bf1   (WMMA GEMM)
__global__ __launch_bounds__(256) void k_fc1(const bf16_t* __restrict__ hn,
                                             const bf16_t* __restrict__ wf1,
                                             const float* __restrict__ bf1,
                                             float* __restrict__ z) {
  int lane = threadIdx.x & 31, wv_id = threadIdx.x >> 5;
  int bm = blockIdx.x >> 3, bn = blockIdx.x & 7;
  int r0 = bm * 128 + wv_id * 16;
  int n0 = bn * 64;
  v8f acc[4];
#pragma unroll
  for (int i = 0; i < 4; ++i) acc[i] = zero8();
  for (int kk = 0; kk < 20; ++kk) {
    int k0 = kk * 32;
    v16bf a = load_a_lds(hn, 640, r0, k0, lane);
#pragma unroll
    for (int nt = 0; nt < 4; ++nt) {
      v16bf bfr = load_bT(wf1, 640, n0 + nt * 16, k0, lane);  // B[k][n]=Wf1[n][k]
      acc[nt] = wmma_bf16(a, bfr, acc[nt]);
    }
  }
  int n = lane & 15, hi8 = (lane & 16) ? 8 : 0;
#pragma unroll
  for (int nt = 0; nt < 4; ++nt) {
    float bb = bf1[n0 + nt * 16 + n];
#pragma unroll
    for (int v = 0; v < 8; ++v)
      z[(size_t)(r0 + v + hi8) * 512 + n0 + nt * 16 + n] = acc[nt][v] + bb;
  }
}

// BN1 + relu + dot(Wf2) + sigmoid, one wave per batch row
__global__ __launch_bounds__(256) void k_fc2(const float* __restrict__ z,
                                             const float* __restrict__ st,
                                             const float* __restrict__ g,
                                             const float* __restrict__ b,
                                             const float* __restrict__ wf2,
                                             const float* __restrict__ bf2,
                                             float* __restrict__ out) {
  int lane = threadIdx.x & 31, wv_id = threadIdx.x >> 5;
  int r = blockIdx.x * 8 + wv_id;
  float p = 0.f;
  float cnt = (float)B_TOT;
  for (int j = lane; j < 512; j += 32) {
    float mean = st[64 + j] / cnt;
    float var  = st[576 + j] / cnt - mean * mean;
    float xn = (z[(size_t)r * 512 + j] - mean) * rsqrtf(var + EPSF) * g[j] + b[j];
    p += fmaxf(xn, 0.f) * wf2[j];
  }
#pragma unroll
  for (int off = 16; off > 0; off >>= 1) p += __shfl_down(p, off);
  if (lane == 0) out[r] = 1.f / (1.f + expf(-(p + bf2[0])));
}

// ------------------------------------------------------------------------- launch
extern "C" void kernel_launch(void* const* d_in, const int* in_sizes, int n_in,
                              void* d_out, int out_size, void* d_ws, size_t ws_size,
                              hipStream_t stream) {
  (void)in_sizes; (void)n_in; (void)out_size; (void)ws_size;
  const float* x      = (const float*)d_in[0];
  const float* bnin_g = (const float*)d_in[1];
  const float* bnin_b = (const float*)d_in[2];
  const float* Wi     = (const float*)d_in[3];
  const float* bi     = (const float*)d_in[4];
  const float* Wq     = (const float*)d_in[5];
  const float* bq     = (const float*)d_in[6];
  const float* Wk     = (const float*)d_in[7];
  const float* bk     = (const float*)d_in[8];
  const float* Wv     = (const float*)d_in[9];
  const float* bv     = (const float*)d_in[10];
  const float* Wg     = (const float*)d_in[11];
  const float* bg     = (const float*)d_in[12];
  const float* Wp     = (const float*)d_in[13];
  const float* bp     = (const float*)d_in[14];
  const float* W1     = (const float*)d_in[15];
  const float* b1     = (const float*)d_in[16];
  const float* W2     = (const float*)d_in[17];
  const float* b2     = (const float*)d_in[18];
  const float* ln1g   = (const float*)d_in[19];
  const float* ln1b   = (const float*)d_in[20];
  const float* ln2g   = (const float*)d_in[21];
  const float* ln2b   = (const float*)d_in[22];
  const float* bno_g  = (const float*)d_in[23];
  const float* bno_b  = (const float*)d_in[24];
  const float* Wf1    = (const float*)d_in[25];
  const float* bf1    = (const float*)d_in[26];
  const float* bn1_g  = (const float*)d_in[27];
  const float* bn1_b  = (const float*)d_in[28];
  const float* Wf2    = (const float*)d_in[29];
  const float* bf2    = (const float*)d_in[30];

  char* ws = (char*)d_ws;
  size_t off = 0;
  float* hbuf = (float*)(ws + off);  off += (size_t)B_TOT * 640 * 4;
  float* z    = (float*)(ws + off);  off += (size_t)B_TOT * 512 * 4;
  float* st   = (float*)(ws + off);  off += 2048 * 4;
  bf16_t* hn  = (bf16_t*)(ws + off); off += (size_t)B_TOT * 640 * 2;
  bf16_t* wqB = (bf16_t*)(ws + off); off += (size_t)L_CNT * H_CNT * 64 * 64 * 2;
  bf16_t* wkB = (bf16_t*)(ws + off); off += (size_t)L_CNT * H_CNT * 64 * 64 * 2;
  bf16_t* wvB = (bf16_t*)(ws + off); off += (size_t)L_CNT * H_CNT * 64 * 64 * 2;
  bf16_t* wpB = (bf16_t*)(ws + off); off += (size_t)L_CNT * 64 * 1024 * 2;
  bf16_t* w1B = (bf16_t*)(ws + off); off += (size_t)L_CNT * 256 * 64 * 2;
  bf16_t* w2B = (bf16_t*)(ws + off); off += (size_t)L_CNT * 64 * 256 * 2;
  bf16_t* wf1B = (bf16_t*)(ws + off); off += (size_t)512 * 640 * 2;

  auto cast = [&](const float* s, bf16_t* d, int n) {
    k_cast<<<(n + 255) / 256, 256, 0, stream>>>(s, d, n);
  };
  cast(Wq, wqB, L_CNT * H_CNT * 64 * 64);
  cast(Wk, wkB, L_CNT * H_CNT * 64 * 64);
  cast(Wv, wvB, L_CNT * H_CNT * 64 * 64);
  cast(Wp, wpB, L_CNT * 64 * 1024);
  cast(W1, w1B, L_CNT * 256 * 64);
  cast(W2, w2B, L_CNT * 64 * 256);
  cast(Wf1, wf1B, 512 * 640);

  k_zero<<<8, 256, 0, stream>>>(st, 2048);
  k_colstats<<<512, 256, 0, stream>>>(x, (long)B_TOT * S_LEN * F_IN, F_IN, S_LEN,
                                      st, st + 16);
  k_embed<<<(B_TOT * 640 + 255) / 256, 256, 0, stream>>>(x, st, bnin_g, bnin_b,
                                                         Wi, bi, hbuf);
  const size_t SMEM = 247040;  // ~241 KB dynamic LDS (gfx1250 WGP: 320 KB)
  k_transformer<<<B_TOT / 16, 256, SMEM, stream>>>(
      hbuf, wqB, wkB, wvB, wpB, w1B, w2B, bq, bk, bv, Wg, bg, bp, b1, b2,
      ln1g, ln1b, ln2g, ln2b);
  k_colstats<<<1024, 256, 0, stream>>>(hbuf, (long)B_TOT * 640, 64, S_LEN,
                                       st + 32, st + 48);
  k_bnout<<<(B_TOT * 640 + 255) / 256, 256, 0, stream>>>(hbuf, st, bno_g, bno_b, hn);
  k_fc1<<<(B_TOT / 128) * (512 / 64), 256, 0, stream>>>(hn, wf1B, bf1, z);
  k_colstats<<<1024, 256, 0, stream>>>(z, (long)B_TOT * 512, 1, 512, st + 64,
                                       st + 576);
  k_fc2<<<B_TOT / 8, 256, 0, stream>>>(z, st, bn1_g, bn1_b, Wf2, bf2, (float*)d_out);
}
